// My_GAT_21698174779815
// MI455X (gfx1250) — compile-verified, matrix operand order checked by reading, not run on hardware
//
#include <hip/hip_runtime.h>
#include <cstdint>
#include <cstddef>

typedef _Float16 h16;
typedef __attribute__((ext_vector_type(16))) _Float16 v16h;
typedef __attribute__((ext_vector_type(8)))  _Float16 v8h;
typedef __attribute__((ext_vector_type(8)))  float    v8f;

#define DD   768
#define BL   4096      // B*L rows
#define NH   12
#define LSEQ 1024
#define NB   4

// ---------------------------------------------------------------------------
// CDNA5 async-to-LDS path (ASYNCcnt), guarded so either toolchain compiles
// ---------------------------------------------------------------------------
#if defined(__has_builtin)
#if __has_builtin(__builtin_amdgcn_global_load_async_to_lds_b128) && \
    __has_builtin(__builtin_amdgcn_s_wait_asynccnt)
#define HAVE_ASYNC 1
#endif
#endif
#ifndef HAVE_ASYNC
#define HAVE_ASYNC 0
#endif

#if HAVE_ASYNC
#define AS1 __attribute__((address_space(1)))
#define AS3 __attribute__((address_space(3)))
typedef int vi4 __attribute__((vector_size(16)));   // GCC-style int4 (matches builtin)
__device__ __forceinline__ void async_ld_b128(const void* g, void* l) {
  __builtin_amdgcn_global_load_async_to_lds_b128((AS1 vi4*)g, (AS3 vi4*)l, 0, 0);
}
__device__ __forceinline__ void wait_async() {
  __builtin_amdgcn_s_wait_asynccnt(0);
}
#else
__device__ __forceinline__ void wait_async() {}
#endif

// ---------------------------------------------------------------------------
// WMMA helpers
// ---------------------------------------------------------------------------
__device__ __forceinline__ v16h load_frag(const h16* base, int stride, int lane) {
  // v_wmma_*_16x16x32_f16 A/B fragment:
  //   lanes 0-15 : row = lane,    K slots {0..7, 16..23}
  //   lanes 16-31: row = lane-16, K slots {8..15, 24..31}
  const int r = lane & 15, hf = lane >> 4;
  const h16* p = base + r * stride + hf * 8;
  v8h lo = *(const v8h*)(p);
  v8h hi = *(const v8h*)(p + 16);
  v16h out;
#pragma unroll
  for (int i = 0; i < 8; ++i) { out[i] = lo[i]; out[i + 8] = hi[i]; }
  return out;
}

__device__ __forceinline__ v8f wmma_f16(v16h a, v16h b, v8f c) {
  return __builtin_amdgcn_wmma_f32_16x16x32_f16(false, a, false, b,
                                                (short)0, c, false, false);
}

// ---------------------------------------------------------------------------
// Setup kernels
// ---------------------------------------------------------------------------
__global__ void transw_kernel(const float* __restrict__ W, h16* __restrict__ Wt,
                              int nmat) {
  long idx = (long)blockIdx.x * blockDim.x + threadIdx.x;
  long total = (long)nmat * DD * DD;
  if (idx >= total) return;
  long mat = idx / ((long)DD * DD);
  int  rem = (int)(idx - mat * (long)DD * DD);
  int  r = rem / DD, c = rem % DD;
  Wt[mat * (long)DD * DD + (long)c * DD + r] = (h16)W[idx];
}

__global__ void packmask_kernel(const int* __restrict__ adj,
                                uint32_t* __restrict__ bits, int nwords) {
  int wi = blockIdx.x * blockDim.x + threadIdx.x;
  if (wi >= nwords) return;
  const int* p = adj + (size_t)wi * 32;
  uint32_t m = 0;
#pragma unroll
  for (int j = 0; j < 32; ++j) m |= (uint32_t)(p[j] != 0) << j;
  bits[wi] = m;
}

__global__ void init_h_kernel(const float* __restrict__ src,
                              float* __restrict__ hf, h16* __restrict__ hp, int n) {
  int i = blockIdx.x * blockDim.x + threadIdx.x;
  if (i >= n) return;
  float v = src[i];
  hf[i] = v; hp[i] = (h16)v;
}

__global__ void combine_kernel(const float* __restrict__ x, const float* __restrict__ y,
                               float* __restrict__ hf, h16* __restrict__ hp, int n) {
  int i = blockIdx.x * blockDim.x + threadIdx.x;
  if (i >= n) return;
  float v = x[i] + y[i];
  v = v > 0.f ? v : 0.f;
  hf[i] = v; hp[i] = (h16)v;
}

__global__ void final_add_kernel(const float* __restrict__ a, const float* __restrict__ b,
                                 float* __restrict__ out, int n) {
  int i = blockIdx.x * blockDim.x + threadIdx.x;
  if (i >= n) return;
  out[i] = a[i] + b[i];
}

// ---------------------------------------------------------------------------
// GEMM: C[4096,768] = alpha * A16 @ W (Bt = W^T, N-major).  Double-buffered
// LDS, tile k+1 streamed (async-to-LDS when available) while computing tile k.
// Block tile 128x128, 8 waves (2x4), wave tile 64x32, WMMA 16x16x32.
// ---------------------------------------------------------------------------
template <bool STORE16>
__global__ __launch_bounds__(256) void gemm_f16(const h16* __restrict__ A,
                                                const h16* __restrict__ Bt,
                                                void* __restrict__ outp,
                                                float alpha) {
  __shared__ h16 As[2][128 * 40];   // 128 rows x 32 K, padded to 40
  __shared__ h16 Bs[2][128 * 40];
  const int m0 = blockIdx.x * 128;
  const int n0 = blockIdx.y * 128;
  const int t = threadIdx.x;
  const int lane = t & 31, w = t >> 5;
  const int wm = w >> 2, wn = w & 3;            // 2 x 4 wave grid
  const int row = t >> 1, half = t & 1;

  const h16* gA = A  + (size_t)(m0 + row) * DD + half * 16;
  const h16* gB = Bt + (size_t)(n0 + row) * DD + half * 16;

#if HAVE_ASYNC
  auto issue = [&](int kt, int buf) {
    h16* dA = &As[buf][row * 40 + half * 16];
    h16* dB = &Bs[buf][row * 40 + half * 16];
    const h16* pA = gA + kt * 32;
    const h16* pB = gB + kt * 32;
    async_ld_b128(pA, dA);     async_ld_b128(pA + 8, dA + 8);
    async_ld_b128(pB, dB);     async_ld_b128(pB + 8, dB + 8);
  };
  issue(0, 0);
#else
  v8h rA0, rA1, rB0, rB1;
  auto fetch = [&](int kt) {
    rA0 = *(const v8h*)(gA + kt * 32); rA1 = *(const v8h*)(gA + kt * 32 + 8);
    rB0 = *(const v8h*)(gB + kt * 32); rB1 = *(const v8h*)(gB + kt * 32 + 8);
  };
  auto stash = [&](int buf) {
    *(v8h*)&As[buf][row * 40 + half * 16]     = rA0;
    *(v8h*)&As[buf][row * 40 + half * 16 + 8] = rA1;
    *(v8h*)&Bs[buf][row * 40 + half * 16]     = rB0;
    *(v8h*)&Bs[buf][row * 40 + half * 16 + 8] = rB1;
  };
  fetch(0); stash(0);
#endif

  v8f acc[4][2];
#pragma unroll
  for (int i = 0; i < 4; ++i)
#pragma unroll
    for (int j = 0; j < 2; ++j) acc[i][j] = (v8f){};

  for (int kt = 0; kt < 24; ++kt) {
    wait_async();
    __syncthreads();
    const int cur = kt & 1, nxt = cur ^ 1;
#if HAVE_ASYNC
    if (kt < 23) issue(kt + 1, nxt);
#else
    if (kt < 23) fetch(kt + 1);
#endif
    v16h a[4], b[2];
#pragma unroll
    for (int i = 0; i < 4; ++i)
      a[i] = load_frag(&As[cur][(wm * 64 + i * 16) * 40], 40, lane);
#pragma unroll
    for (int j = 0; j < 2; ++j)
      b[j] = load_frag(&Bs[cur][(wn * 32 + j * 16) * 40], 40, lane);
#pragma unroll
    for (int i = 0; i < 4; ++i)
#pragma unroll
      for (int j = 0; j < 2; ++j)
        acc[i][j] = wmma_f16(a[i], b[j], acc[i][j]);
#if !HAVE_ASYNC
    if (kt < 23) stash(nxt);
#endif
  }

  const int lane15 = lane & 15, hf = lane >> 4;
#pragma unroll
  for (int i = 0; i < 4; ++i)
#pragma unroll
    for (int j = 0; j < 2; ++j)
#pragma unroll
      for (int r = 0; r < 8; ++r) {
        int orow = m0 + wm * 64 + i * 16 + r + 8 * hf;
        int col  = n0 + wn * 32 + j * 16 + lane15;
        float v = acc[i][j][r] * alpha;
        if (STORE16) ((h16*)outp)[(size_t)orow * DD + col] = (h16)v;
        else         ((float*)outp)[(size_t)orow * DD + col] = v;
      }
}

// ---------------------------------------------------------------------------
// Flash attention per (b, head, 64-query tile). 4 waves x 16 q rows.
// Double-buffered K/V tiles; K streamed async-to-LDS, V register-staged
// (needs transpose). Q pre-scaled by 1/sqrt(dk). Packed-bit mask.
// ---------------------------------------------------------------------------
__global__ __launch_bounds__(128) void attn_kernel(const h16* __restrict__ Q,
                                                   const h16* __restrict__ K,
                                                   const h16* __restrict__ V,
                                                   h16* __restrict__ O,
                                                   const uint32_t* __restrict__ mbits) {
  __shared__ h16 Ks[2][64 * 72];     // [key][d]
  __shared__ h16 Vt[2][64 * 72];     // [d][key]
  __shared__ h16 Ps[4 * 16 * 72];    // per-wave P patch

  const int b = blockIdx.y / NH, h = blockIdx.y % NH;
  const int qt = blockIdx.x;
  const int t = threadIdx.x, lane = t & 31, w = t >> 5;
  const int lane15 = lane & 15, hf = lane >> 4;
  const int qbase = qt * 64 + w * 16;
  const int row = t >> 1, part = t & 1;
  const size_t tstep = (size_t)64 * DD;

  const h16* gK = K + (size_t)(b * LSEQ + row) * DD + h * 64 + part * 32;
  const h16* gV = V + (size_t)(b * LSEQ + row) * DD + h * 64 + part * 32;

  v8h rv[4];
  auto fetchV = [&](int kt) {
#pragma unroll
    for (int q4 = 0; q4 < 4; ++q4) rv[q4] = *(const v8h*)(gV + kt * tstep + q4 * 8);
  };
  auto stashV = [&](int buf) {
#pragma unroll
    for (int q4 = 0; q4 < 4; ++q4)
#pragma unroll
      for (int e = 0; e < 8; ++e)
        Vt[buf][(part * 32 + q4 * 8 + e) * 72 + row] = rv[q4][e];
  };
#if HAVE_ASYNC
  auto issueK = [&](int kt, int buf) {
    h16* d = &Ks[buf][row * 72 + part * 32];
    const h16* p = gK + kt * tstep;
#pragma unroll
    for (int q4 = 0; q4 < 4; ++q4) async_ld_b128(p + q4 * 8, d + q4 * 8);
  };
  issueK(0, 0);
#else
  v8h rk[4];
  auto fetchK = [&](int kt) {
#pragma unroll
    for (int q4 = 0; q4 < 4; ++q4) rk[q4] = *(const v8h*)(gK + kt * tstep + q4 * 8);
  };
  auto stashK = [&](int buf) {
#pragma unroll
    for (int q4 = 0; q4 < 4; ++q4)
      *(v8h*)&Ks[buf][row * 72 + part * 32 + q4 * 8] = rk[q4];
  };
  fetchK(0); stashK(0);
#endif
  fetchV(0); stashV(0);

  // Q fragments: 16 rows x 64 d, two 32-wide K chunks
  v16h aQ[2];
#pragma unroll
  for (int c = 0; c < 2; ++c)
    aQ[c] = load_frag(Q + (size_t)(b * LSEQ + qbase) * DD + h * 64 + c * 32,
                      DD, lane);

  v8f o[4];
#pragma unroll
  for (int j = 0; j < 4; ++j) o[j] = (v8f){};
  float m[8], l[8];
#pragma unroll
  for (int r = 0; r < 8; ++r) { m[r] = -1e30f; l[r] = 0.f; }

  for (int kt = 0; kt < LSEQ / 64; ++kt) {
    wait_async();
    __syncthreads();
    const int cur = kt & 1, nxt = cur ^ 1;
    const bool more = kt < LSEQ / 64 - 1;
#if HAVE_ASYNC
    if (more) issueK(kt + 1, nxt);
#else
    if (more) fetchK(kt + 1);
#endif
    if (more) fetchV(kt + 1);

    // S = Q @ K^T  (4 key tiles of 16)
    v8f s[4];
#pragma unroll
    for (int j = 0; j < 4; ++j) s[j] = (v8f){};
#pragma unroll
    for (int c = 0; c < 2; ++c)
#pragma unroll
      for (int j = 0; j < 4; ++j) {
        v16h bK = load_frag(&Ks[cur][(j * 16) * 72 + c * 32], 72, lane);
        s[j] = wmma_f16(aQ[c], bK, s[j]);
      }

    // mask + online softmax (C layout: VGPR r, lane -> row r+8*hf, col lane15)
#pragma unroll
    for (int r = 0; r < 8; ++r) {
      int qrow = qbase + r + 8 * hf;
      uint2 mw = *(const uint2*)(mbits + (size_t)(b * LSEQ + qrow) * (LSEQ / 32) + kt * 2);
      float sv[4];
#pragma unroll
      for (int j = 0; j < 4; ++j) {
        int cbit = (j & 1) * 16 + lane15;
        uint32_t wd = (j < 2) ? mw.x : mw.y;
        sv[j] = ((wd >> cbit) & 1u) ? s[j][r] : -1e9f;
      }
      float rm = fmaxf(fmaxf(sv[0], sv[1]), fmaxf(sv[2], sv[3]));
#pragma unroll
      for (int off = 1; off < 16; off <<= 1)
        rm = fmaxf(rm, __shfl_xor(rm, off));
      float nm = fmaxf(m[r], rm);
      float sc = __expf(m[r] - nm);
      m[r] = nm;
      float ps = 0.f;
#pragma unroll
      for (int j = 0; j < 4; ++j) {
        float p = __expf(sv[j] - nm);
        ps += p;
        Ps[(w * 16 + r + 8 * hf) * 72 + j * 16 + lane15] = (h16)p;
      }
#pragma unroll
      for (int off = 1; off < 16; off <<= 1)
        ps += __shfl_xor(ps, off);
      l[r] = l[r] * sc + ps;
#pragma unroll
      for (int j = 0; j < 4; ++j) o[j][r] *= sc;
    }

    // O += P @ V   (A = P from per-wave LDS patch, B = V^T)
#pragma unroll
    for (int c = 0; c < 2; ++c) {
      v16h aP = load_frag(&Ps[(w * 16) * 72 + c * 32], 72, lane);
#pragma unroll
      for (int j = 0; j < 4; ++j) {
        v16h bV = load_frag(&Vt[cur][(j * 16) * 72 + c * 32], 72, lane);
        o[j] = wmma_f16(aP, bV, o[j]);
      }
    }
    if (more) stashV(nxt);
#if !HAVE_ASYNC
    if (more) stashK(nxt);
#endif
  }

#pragma unroll
  for (int j = 0; j < 4; ++j)
#pragma unroll
    for (int r = 0; r < 8; ++r) {
      int qrow = qbase + r + 8 * hf;
      int col = h * 64 + j * 16 + lane15;
      O[(size_t)(b * LSEQ + qrow) * DD + col] = (h16)(o[j][r] / l[r]);
    }
}

// ---------------------------------------------------------------------------
// Residual + LayerNorm: out = LN(G + X) * gamma + beta, one block per row
// ---------------------------------------------------------------------------
__global__ __launch_bounds__(256) void ln_kernel(const float* __restrict__ G,
                                                 const float* __restrict__ X,
                                                 const float* __restrict__ gam,
                                                 const float* __restrict__ bet,
                                                 float* __restrict__ out) {
  __shared__ float red[16];
  const int row = blockIdx.x;
  const int t = threadIdx.x, lane = t & 31, w = t >> 5;
  float v[3], s = 0.f, s2 = 0.f;
#pragma unroll
  for (int e = 0; e < 3; ++e) {
    int c = t + e * 256;
    float x = G[(size_t)row * DD + c] + X[(size_t)row * DD + c];
    v[e] = x; s += x; s2 += x * x;
  }
#pragma unroll
  for (int off = 1; off < 32; off <<= 1) {
    s += __shfl_xor(s, off); s2 += __shfl_xor(s2, off);
  }
  if (lane == 0) { red[w] = s; red[8 + w] = s2; }
  __syncthreads();
  float ts = 0.f, ts2 = 0.f;
#pragma unroll
  for (int i = 0; i < 8; ++i) { ts += red[i]; ts2 += red[8 + i]; }
  float mu = ts * (1.f / DD);
  float var = ts2 * (1.f / DD) - mu * mu;
  float inv = rsqrtf(var + 1e-6f);
#pragma unroll
  for (int e = 0; e < 3; ++e) {
    int c = t + e * 256;
    out[(size_t)row * DD + c] = (v[e] - mu) * inv * gam[c] + bet[c];
  }
}

// ---------------------------------------------------------------------------
// Host orchestration
// ---------------------------------------------------------------------------
extern "C" void kernel_launch(void* const* d_in, const int* in_sizes, int n_in,
                              void* d_out, int out_size, void* d_ws, size_t ws_size,
                              hipStream_t stream) {
  const float* h_a  = (const float*)d_in[0];
  const float* h_b  = (const float*)d_in[1];
  const int* adj[4] = {(const int*)d_in[2], (const int*)d_in[3],
                       (const int*)d_in[4], (const int*)d_in[5]};
  const float* Wmat[4] = {(const float*)d_in[6], (const float*)d_in[7],
                          (const float*)d_in[8], (const float*)d_in[9]};
  const float* ln_g = (const float*)d_in[10];
  const float* ln_b = (const float*)d_in[11];

  char* ws = (char*)d_ws;
  const size_t SZW = (size_t)DD * DD;                 // elems per weight matrix
  h16*      wt   = (h16*)(ws + 0);                    // 32 mats, kind-major
  uint32_t* mb   = (uint32_t*)(ws + 37748736);        // 4 x 131072 words
  h16*      ha16 = (h16*)(ws + 39845888);
  h16*      hb16 = (h16*)(ws + 46137344);
  h16*      q16  = (h16*)(ws + 52428800);
  h16*      k16  = (h16*)(ws + 58720256);
  h16*      vv16 = (h16*)(ws + 65011712);
  h16*      o16  = (h16*)(ws + 71303168);
  float*    g32  = (float*)(ws + 77594624);
  float*    modb = (float*)(ws + 90177536);           // 4 x 12582912 B
  float*    haf  = (float*)(ws + 140509184);
  float*    hbf  = (float*)(ws + 153092096);

  const int NACT = BL * DD;                           // 3145728
  const int MWORDS = NB * LSEQ * LSEQ / 32;           // 131072 per mask

  for (int kind = 0; kind < 4; ++kind)
    transw_kernel<<<(int)((8 * SZW + 255) / 256), 256, 0, stream>>>(
        Wmat[kind], wt + (size_t)kind * 8 * SZW, 8);
  for (int mk = 0; mk < 4; ++mk)
    packmask_kernel<<<(MWORDS + 255) / 256, 256, 0, stream>>>(
        adj[mk], mb + (size_t)mk * MWORDS, MWORDS);
  init_h_kernel<<<(NACT + 255) / 256, 256, 0, stream>>>(h_a, haf, ha16, NACT);
  init_h_kernel<<<(NACT + 255) / 256, 256, 0, stream>>>(h_b, hbf, hb16, NACT);

  for (int i = 0; i < 2; ++i) {
    // module order: 0=a2a, 1=b2b, 2=a2b(q=h_b,k=h_a), 3=b2a(q=h_a,k=h_b)
    const h16*   xq16[4] = {ha16, hb16, hb16, ha16};
    const h16*   xk16[4] = {ha16, hb16, ha16, hb16};
    const float* xqf [4] = {haf,  hbf,  hbf,  haf };
    for (int mdl = 0; mdl < 4; ++mdl) {
      int mi = i * 4 + mdl;
      const uint32_t* mask = mb + (size_t)mdl * MWORDS;
      float* modout = modb + (size_t)mdl * NACT;
      gemm_f16<true><<<dim3(32, 6), 256, 0, stream>>>(
          xq16[mdl], wt + (0 * 8 + mi) * SZW, (void*)q16, 0.125f);
      gemm_f16<true><<<dim3(32, 6), 256, 0, stream>>>(
          xk16[mdl], wt + (1 * 8 + mi) * SZW, (void*)k16, 1.0f);
      gemm_f16<true><<<dim3(32, 6), 256, 0, stream>>>(
          xk16[mdl], wt + (2 * 8 + mi) * SZW, (void*)vv16, 1.0f);
      attn_kernel<<<dim3(LSEQ / 64, NB * NH), 128, 0, stream>>>(q16, k16, vv16,
                                                                o16, mask);
      gemm_f16<false><<<dim3(32, 6), 256, 0, stream>>>(
          o16, wt + (3 * 8 + mi) * SZW, (void*)g32, 1.0f);
      ln_kernel<<<BL, 256, 0, stream>>>(g32, xqf[mdl], ln_g + (size_t)mi * DD,
                                        ln_b + (size_t)mi * DD, modout);
    }
    combine_kernel<<<(NACT + 255) / 256, 256, 0, stream>>>(
        modb + 0 * (size_t)NACT, modb + 3 * (size_t)NACT, haf, ha16, NACT);
    combine_kernel<<<(NACT + 255) / 256, 256, 0, stream>>>(
        modb + 1 * (size_t)NACT, modb + 2 * (size_t)NACT, hbf, hb16, NACT);
  }

  float* out = (float*)d_out;
  final_add_kernel<<<(NACT + 255) / 256, 256, 0, stream>>>(h_a, haf, out, NACT);
  final_add_kernel<<<(NACT + 255) / 256, 256, 0, stream>>>(h_b, hbf, out + NACT, NACT);
  (void)in_sizes; (void)n_in; (void)out_size; (void)ws_size;
}
